// RationalBasicBlock_29094108463587
// MI455X (gfx1250) — matrix operand order repaired and checked by reading, not compiled, not run on hardware
//
#include <hip/hip_runtime.h>

// ---------------------------------------------------------------------------
// RationalBasicBlock on MI455X (gfx1250):
//   conv3x3 -> BN -> Pade  -> conv3x3 -> BN -> +x -> Pade
// Implicit-GEMM with V_WMMA_F32_16X16X32_BF16 and bf16(hi)+bf16(lo) error
// compensation (3 WMMAs per K-step) for ~fp32 accuracy at bf16 matrix rate.
// Slab staging uses all three CDNA5 copy paths in parallel:
//   TDM TENSOR_LOAD_TO_LDS (hi slab interior, TENSORcnt, LDS-padding in D#),
//   GLOBAL_LOAD_ASYNC_TO_LDS_B128 (lo slab interior, ASYNCcnt),
//   vector ds_store zero-fill (halo ring, DScnt).
// ---------------------------------------------------------------------------

typedef __attribute__((ext_vector_type(16))) __bf16 v16bf;
typedef __attribute__((ext_vector_type(8)))  float  v8f;
typedef __attribute__((ext_vector_type(4)))  unsigned v4u;
typedef __attribute__((ext_vector_type(8)))  int      v8i;
typedef __attribute__((ext_vector_type(4)))  int      v4i;

union BF16Frag { uint4 q[2]; v16bf v; };

__device__ __forceinline__ unsigned short f2bf(float f) {
  unsigned u = __float_as_uint(f);
  u += 0x7FFFu + ((u >> 16) & 1u);          // round-to-nearest-even
  return (unsigned short)(u >> 16);
}
__device__ __forceinline__ float bf2f(unsigned short h) {
  return __uint_as_float(((unsigned)h) << 16);
}
__device__ __forceinline__ v8f zero8() {
  v8f z;
#pragma unroll
  for (int i = 0; i < 8; ++i) z[i] = 0.0f;
  return z;
}

#define BN_   32
#define CC_   256
#define HH_   56
#define WW_   56
#define HW_   3136
#define KSTEPS 72          // 3*3 * (256/32)
#define SLAB_W 68          // 66 used (w' = -1..64), padded
#define SLAB_CI 264        // 256 ci + 8 pad (bank-conflict-free b128 reads)
#define SLAB_ROW (SLAB_W * SLAB_CI)
#define SLAB_ELEMS (3 * SLAB_ROW)          // 53856 bf16 per (hi|lo)
#define CONV_LDS_BYTES (2 * SLAB_ELEMS * 2)  // 215424 B < 320 KB WGP LDS

// ---------------------------------------------------------------------------
// TDM: load a 2D tile [56 rows x 256 bf16] (row stride 256 elems in memory)
// into LDS with 8-element padding after each row (264-elem LDS stride).
// D# built per CDNA5 ISA 8.3/8.4; trailing groups zero => dims 2+ unused (2D).
// ---------------------------------------------------------------------------
__device__ __forceinline__ void tdm_load_row_2d(unsigned lds_addr,
                                                unsigned long long gaddr) {
  v4u g0;
  g0[0] = 1u;                                           // count=1 (valid), user mode
  g0[1] = lds_addr;                                     // lds_addr [63:32]
  g0[2] = (unsigned)(gaddr & 0xFFFFFFFFull);            // global_addr [95:64]
  g0[3] = (unsigned)((gaddr >> 32) & 0x1FFFFFFull)      // global_addr [120:96]
        | (2u << 30);                                   // type=2 ("image") [127:126]
  v8i g1;
  g1[0] = (int)((1u << 16)      // data_size = 1 -> 2 bytes
              | (1u << 20)      // pad_enable
              | (6u << 22)      // pad_interval = 6 -> every 128 DWORDs (256 bf16)
              | (3u << 25));    // pad_amount  = 3 -> 4 DWORDs (8 bf16)
  g1[1] = (int)(256u << 16);    // tensor_dim0 = 256   (bits 79:48)
  g1[2] = (int)(56u << 16);     // tensor_dim1 = 56    (bits 111:80)
  g1[3] = (int)(256u << 16);    // tile_dim0   = 256   (bits 127:112)
  g1[4] = (int)56;              // tile_dim1   = 56; tile_dim2 = 0 (2D)
  g1[5] = (int)256;             // tensor_dim0_stride = 256 (bits 207:160)
  g1[6] = (int)(256u << 16);    // tensor_dim1_stride = 256 (bits 255:208)
  g1[7] = 0;
  v4i z4 = {0, 0, 0, 0};
  v8i z8 = {0, 0, 0, 0, 0, 0, 0, 0};
  __builtin_amdgcn_tensor_load_to_lds(g0, g1, z4, z4, z8, 0);
}

// ---------------------------------------------------------------------------
// BN folding: scale = gamma*rsqrt(var+eps), bias = beta - mean*scale
// ---------------------------------------------------------------------------
__global__ void bn_prep(const float* g1, const float* b1, const float* m1, const float* v1,
                        const float* g2, const float* b2, const float* m2, const float* v2,
                        float* s1, float* o1, float* s2, float* o2) {
  int c = threadIdx.x;
  float i1 = g1[c] * rsqrtf(v1[c] + 1e-5f);
  s1[c] = i1; o1[c] = b1[c] - m1[c] * i1;
  float i2 = g2[c] * rsqrtf(v2[c] + 1e-5f);
  s2[c] = i2; o2[c] = b2[c] - m2[c] * i2;
}

// ---------------------------------------------------------------------------
// Pack OIHW fp32 weights into WMMA A-fragment order (16x32 bf16 tiles),
// hi/lo split.  K-step order: ks = (kh*3+kw)*8 + cib.
// Layout: [ks][mtile][lane][16 bf16]  (lane-major, 32 B per lane -> coalesced)
// Lane l<16 holds M=l, K = {0..15}; lanes 16-31 hold M=l-16, K={16..31}
// ---------------------------------------------------------------------------
__global__ void pack_weights(const float* __restrict__ w,
                             unsigned short* __restrict__ ahi,
                             unsigned short* __restrict__ alo) {
  int g = blockIdx.x * 256 + threadIdx.x;    // 72*16*32 = 36864 threads
  int lane = g & 31;
  int mt   = (g >> 5) & 15;
  int ks   = g >> 9;
  int half = lane >> 4;
  int m    = lane & 15;
  int co   = mt * 16 + m;
  int kh = ks / 24, kw = (ks / 8) % 3, cib = ks & 7;

  unsigned int hq[8], lq[8];
  unsigned short hprev = 0, lprev = 0;
#pragma unroll
  for (int j = 0; j < 16; ++j) {
    int kk = 16 * half + j;
    int ci = cib * 32 + kk;
    float val = w[((size_t)(co * 256 + ci) * 3 + kh) * 3 + kw];
    unsigned short hb = f2bf(val);
    unsigned short lb = f2bf(val - bf2f(hb));
    if (j & 1) {
      hq[j >> 1] = (unsigned)hprev | ((unsigned)hb << 16);
      lq[j >> 1] = (unsigned)lprev | ((unsigned)lb << 16);
    } else { hprev = hb; lprev = lb; }
  }
  size_t base = (size_t)g * 16;
  *(uint4*)(ahi + base)     = make_uint4(hq[0], hq[1], hq[2], hq[3]);
  *(uint4*)(ahi + base + 8) = make_uint4(hq[4], hq[5], hq[6], hq[7]);
  *(uint4*)(alo + base)     = make_uint4(lq[0], lq[1], lq[2], lq[3]);
  *(uint4*)(alo + base + 8) = make_uint4(lq[4], lq[5], lq[6], lq[7]);
}

// ---------------------------------------------------------------------------
// NCHW fp32 -> NHWC bf16 hi/lo, LDS-transposed so both sides are coalesced.
// Block: 64 ci x 64 hw tile of one image.
// ---------------------------------------------------------------------------
__global__ void nchw_to_nhwc_bf16(const float* __restrict__ x,
                                  unsigned short* __restrict__ xhi,
                                  unsigned short* __restrict__ xlo) {
  __shared__ float tile[64][65];
  int n = blockIdx.z, cit = blockIdx.y, hwt = blockIdx.x;
  int t = threadIdx.x;
  int col = t & 63, rg = t >> 6;
#pragma unroll
  for (int k = 0; k < 16; ++k) {
    int ci = rg * 16 + k;
    tile[ci][col] = x[(size_t)(n * CC_ + cit * 64 + ci) * HW_ + hwt * 64 + col];
  }
  __syncthreads();
  int hw = t >> 2, cg = t & 3;
  alignas(16) unsigned short hs[16], ls[16];
#pragma unroll
  for (int k = 0; k < 16; ++k) {
    float v = tile[cg * 16 + k][hw];
    unsigned short hb = f2bf(v);
    hs[k] = hb;
    ls[k] = f2bf(v - bf2f(hb));
  }
  size_t o = (size_t)(n * HW_ + hwt * 64 + hw) * CC_ + cit * 64 + cg * 16;
  *(uint4*)(xhi + o)     = *(const uint4*)hs;
  *(uint4*)(xhi + o + 8) = *(const uint4*)(hs + 8);
  *(uint4*)(xlo + o)     = *(const uint4*)ls;
  *(uint4*)(xlo + o + 8) = *(const uint4*)(ls + 8);
}

// ---------------------------------------------------------------------------
// Fused conv3x3 + BN + Pade (+ residual for SECOND).
// One block = one (n,h) output row: M=256 (all co), N=64 (56 valid w).
// 8 waves; wave computes 32(co) x 64(w) = 8 WMMA tiles.
// ---------------------------------------------------------------------------
template <bool SECOND>
__global__ __launch_bounds__(256, 1)
void conv_wmma(const unsigned short* __restrict__ inhi,
               const unsigned short* __restrict__ inlo,
               const unsigned short* __restrict__ ahi,
               const unsigned short* __restrict__ alo,
               const float* __restrict__ bnscale,
               const float* __restrict__ bnbias,
               const float* __restrict__ acoef,
               const float* __restrict__ bcoef,
               const float* __restrict__ xres,
               unsigned short* __restrict__ outhi,
               unsigned short* __restrict__ outlo,
               float* __restrict__ outf) {
  extern __shared__ char smem[];
  unsigned short* sbhi = (unsigned short*)smem;
  unsigned short* sblo = sbhi + SLAB_ELEMS;

  const int tid = threadIdx.x;
  const int n = blockIdx.x / HH_;
  const int h = blockIdx.x % HH_;

  // ---- Phase 1a: wave 0 issues TDM loads for the hi-slab interior ----
  // One 2D descriptor per valid image row: [56 w][256 ci] -> LDS w'=1..56,
  // with D# padding producing the 264-element LDS row stride.
  if (tid < 32) {
#pragma unroll
    for (int r = 0; r < 3; ++r) {
      int mrow = h + r - 1;
      if (mrow >= 0 && mrow < HH_) {
        unsigned ldsa = (unsigned)(size_t)(sbhi + (r * SLAB_W + 1) * SLAB_CI);
        unsigned long long ga = (unsigned long long)(size_t)inhi +
                                (unsigned long long)((n * HH_ + mrow) * WW_) * (CC_ * 2);
        tdm_load_row_2d(ldsa, ga);
      }
    }
  }

  // ---- Phase 1b: lo-slab interior via async-to-LDS; halo ring zeroed ----
  for (int idx = tid; idx < 3 * 66 * 32; idx += 256) {
    int c8 = idx & 31;
    int wp = (idx >> 5) % 66;
    int r  = idx / (66 * 32);
    int ci0 = c8 * 8;
    int mrow = h + r - 1;
    int mw = wp - 1;
    int so = (r * SLAB_W + wp) * SLAB_CI + ci0;
    if (mrow >= 0 && mrow < HH_ && mw >= 0 && mw < WW_) {
      unsigned goff = (unsigned)(((unsigned)((n * HH_ + mrow) * WW_ + mw) * CC_ + ci0) * 2u);
      unsigned ldsl = (unsigned)(size_t)(sblo + so);   // generic low 32 = LDS addr
      asm volatile("global_load_async_to_lds_b128 %0, %1, %2"
                   :: "v"(ldsl), "v"(goff), "s"(inlo) : "memory");
    } else {
      uint4 z = make_uint4(0u, 0u, 0u, 0u);
      *(uint4*)(sbhi + so) = z;
      *(uint4*)(sblo + so) = z;
    }
  }
  asm volatile("s_wait_asynccnt 0x0" ::: "memory");    // drain this wave's async copies
  if (tid < 32) __builtin_amdgcn_s_wait_tensorcnt(0);  // wave 0 drains TDM
  __syncthreads();

  const int lane = tid & 31;
  const int wv   = tid >> 5;
  const int half = lane >> 4;
  const int lcol = lane & 15;

  v8f acc[2][4];
#pragma unroll
  for (int mi = 0; mi < 2; ++mi)
#pragma unroll
    for (int ni = 0; ni < 4; ++ni) acc[mi][ni] = zero8();

  const size_t alanebase = (size_t)lane * 16;

  // ---- Phase 2: 72 K-steps x 24 WMMAs (bf16x3 compensation) ----
  for (int ks = 0; ks < KSTEPS; ++ks) {
    int kh = ks / 24, kw = (ks / 8) % 3, cib = ks & 7;

    BF16Frag Ah[2], Al[2];
#pragma unroll
    for (int mi = 0; mi < 2; ++mi) {
      size_t ab = (size_t)(ks * 16 + wv * 2 + mi) * 512 + alanebase;
      Ah[mi].q[0] = *(const uint4*)(ahi + ab);
      Ah[mi].q[1] = *(const uint4*)(ahi + ab + 8);
      Al[mi].q[0] = *(const uint4*)(alo + ab);
      Al[mi].q[1] = *(const uint4*)(alo + ab + 8);
    }
    if (ks + 1 < KSTEPS) {
      size_t abn = (size_t)((ks + 1) * 16 + wv * 2) * 512 + alanebase;
      __builtin_prefetch(ahi + abn, 0, 1);   // global_prefetch_b8
      __builtin_prefetch(alo + abn, 0, 1);
    }

    int b0 = (kh * SLAB_W + lcol + kw) * SLAB_CI + cib * 32 + 16 * half;
#pragma unroll
    for (int ni = 0; ni < 4; ++ni) {
      int bo = b0 + ni * 16 * SLAB_CI;
      BF16Frag Bh, Bl;
      Bh.q[0] = *(const uint4*)(sbhi + bo);
      Bh.q[1] = *(const uint4*)(sbhi + bo + 8);
      Bl.q[0] = *(const uint4*)(sblo + bo);
      Bl.q[1] = *(const uint4*)(sblo + bo + 8);
#pragma unroll
      for (int mi = 0; mi < 2; ++mi) {
        acc[mi][ni] = __builtin_amdgcn_wmma_f32_16x16x32_bf16(
            false, Ah[mi].v, false, Bh.v, (short)0, acc[mi][ni], false, false);
        acc[mi][ni] = __builtin_amdgcn_wmma_f32_16x16x32_bf16(
            false, Ah[mi].v, false, Bl.v, (short)0, acc[mi][ni], false, false);
        acc[mi][ni] = __builtin_amdgcn_wmma_f32_16x16x32_bf16(
            false, Al[mi].v, false, Bh.v, (short)0, acc[mi][ni], false, false);
      }
    }
  }

  // ---- Phase 3: epilogue (BN + [residual] + Pade, store) ----
#pragma unroll
  for (int mi = 0; mi < 2; ++mi) {
    int ctile = wv * 32 + mi * 16;
    int grp = ctile >> 6;                       // 64 channels per group (G=4)
    float A0 = acoef[grp * 6 + 0], A1 = acoef[grp * 6 + 1], A2 = acoef[grp * 6 + 2];
    float A3 = acoef[grp * 6 + 3], A4 = acoef[grp * 6 + 4], A5 = acoef[grp * 6 + 5];
    float B0 = bcoef[grp * 4 + 0], B1 = bcoef[grp * 4 + 1];
    float B2 = bcoef[grp * 4 + 2], B3 = bcoef[grp * 4 + 3];
    int cobase = ctile + 8 * half;
    float bs[8], bb[8];
#pragma unroll
    for (int v = 0; v < 8; ++v) { bs[v] = bnscale[cobase + v]; bb[v] = bnbias[cobase + v]; }

#pragma unroll
    for (int ni = 0; ni < 4; ++ni) {
      int w = ni * 16 + lcol;
      bool wok = (w < WW_);
      float vals[8];
#pragma unroll
      for (int v = 0; v < 8; ++v) {
        float y = acc[mi][ni][v] * bs[v] + bb[v];
        if (SECOND && wok)
          y += xres[(size_t)(n * CC_ + cobase + v) * HW_ + h * WW_ + w];
        float P = A5; P = P * y + A4; P = P * y + A3; P = P * y + A2; P = P * y + A1; P = P * y + A0;
        float S = B3; S = S * y + B2; S = S * y + B1; S = S * y + B0;
        float Q = 1.0f + fabsf(S * y);
        vals[v] = P / Q;
      }
      if (wok) {
        if (SECOND) {
#pragma unroll
          for (int v = 0; v < 8; ++v)
            outf[(size_t)(n * CC_ + cobase + v) * HW_ + h * WW_ + w] = vals[v];
        } else {
          alignas(16) unsigned short hs[8], ls[8];
#pragma unroll
          for (int v = 0; v < 8; ++v) {
            unsigned short hb = f2bf(vals[v]);
            hs[v] = hb;
            ls[v] = f2bf(vals[v] - bf2f(hb));
          }
          size_t o = (size_t)((n * HH_ + h) * WW_ + w) * CC_ + cobase;
          *(uint4*)(outhi + o) = *(const uint4*)hs;
          *(uint4*)(outlo + o) = *(const uint4*)ls;
        }
      }
    }
  }
}

// ---------------------------------------------------------------------------
extern "C" void kernel_launch(void* const* d_in, const int* in_sizes, int n_in,
                              void* d_out, int out_size, void* d_ws, size_t ws_size,
                              hipStream_t stream) {
  (void)in_sizes; (void)n_in; (void)out_size; (void)ws_size;
  const float* x  = (const float*)d_in[0];
  const float* w1 = (const float*)d_in[1];
  const float* w2 = (const float*)d_in[2];
  const float* gamma1 = (const float*)d_in[3];
  const float* beta1  = (const float*)d_in[4];
  const float* mean1  = (const float*)d_in[5];
  const float* var1   = (const float*)d_in[6];
  const float* gamma2 = (const float*)d_in[7];
  const float* beta2  = (const float*)d_in[8];
  const float* mean2  = (const float*)d_in[9];
  const float* var2   = (const float*)d_in[10];
  const float* a1 = (const float*)d_in[11];
  const float* b1 = (const float*)d_in[12];
  const float* a2 = (const float*)d_in[13];
  const float* b2 = (const float*)d_in[14];

  char* ws = (char*)d_ws;
  const size_t SZ  = (size_t)BN_ * HH_ * WW_ * CC_;  // activation elements
  const size_t SZB = SZ * 2;                         // bf16 bytes
  const size_t APE = (size_t)KSTEPS * 16 * 512;      // packed-A elems per array

  unsigned short* xhi  = (unsigned short*)(ws);
  unsigned short* xlo  = (unsigned short*)(ws + SZB);
  unsigned short* y1hi = (unsigned short*)(ws + 2 * SZB);
  unsigned short* y1lo = (unsigned short*)(ws + 3 * SZB);
  unsigned short* a1h  = (unsigned short*)(ws + 4 * SZB);
  unsigned short* a1l  = a1h + APE;
  unsigned short* a2h  = a1l + APE;
  unsigned short* a2l  = a2h + APE;
  float* bns1 = (float*)(a2l + APE);
  float* bnb1 = bns1 + 256;
  float* bns2 = bnb1 + 256;
  float* bnb2 = bns2 + 256;

  bn_prep<<<1, 256, 0, stream>>>(gamma1, beta1, mean1, var1,
                                 gamma2, beta2, mean2, var2,
                                 bns1, bnb1, bns2, bnb2);
  pack_weights<<<144, 256, 0, stream>>>(w1, a1h, a1l);
  pack_weights<<<144, 256, 0, stream>>>(w2, a2h, a2l);
  nchw_to_nhwc_bf16<<<dim3(49, 4, 32), 256, 0, stream>>>(x, xhi, xlo);

  conv_wmma<false><<<BN_ * HH_, 256, CONV_LDS_BYTES, stream>>>(
      xhi, xlo, a1h, a1l, bns1, bnb1, a1, b1, nullptr, y1hi, y1lo, nullptr);
  conv_wmma<true><<<BN_ * HH_, 256, CONV_LDS_BYTES, stream>>>(
      y1hi, y1lo, a2h, a2l, bns2, bnb2, a2, b2, x, nullptr, nullptr, (float*)d_out);
}
